// DynamicDecoder_87179246174990
// MI455X (gfx1250) — matrix operand order repaired and checked by reading, not compiled, not run on hardware
//
#include <hip/hip_runtime.h>

// ---------------------------------------------------------------------------
// DynamicDecoder on MI455X (gfx1250): bf16 WMMA GEMMs + in-register GRU gates
// ---------------------------------------------------------------------------

typedef __attribute__((ext_vector_type(16))) __bf16 v16bf;
typedef __attribute__((ext_vector_type(8)))  float  v8f;
typedef __attribute__((ext_vector_type(4)))  unsigned int v4u;

#define DG   150
#define DS   50
#define DIN  12
#define MTILE 64

// Padded bf16 weight panels in d_ws (offsets in halfwords)
#define OFF_R    0        // [160][192]  fused W_ih|W_hh, r gate
#define OFF_Z    30720    // [160][192]  fused W_ih|W_hh, z gate
#define OFF_NIH  61440    // [160][32]   W_ih, n gate
#define OFF_NHH  66560    // [160][160]  W_hh, n gate
#define OFF_W1   92160    // [64][224]   W1
#define OFF_W2   106496   // [64][64]    W2
#define WTOT     110592

// LDS layout (bytes)
#define SA1_B  0          // [64][192] bf16 : A1 = [x_flat | h]
#define SA2_B  24576      // [64][224] bf16 : A2 = [h_new | static]
#define SA3_B  53248      // [64][64]  bf16 : A3 = softplus(layer1)
#define SH_B   61440      // [64][64]  f32  : tanh(layer2)
#define SO6_B  77824      // [64][8]   f32  : head 6-vector
#define SMEM_BYTES 79872

__device__ __forceinline__ unsigned short f2bf(float f) {
  union { float f; unsigned u; } c; c.f = f;
  unsigned u = c.u;
  u += 0x7FFFu + ((u >> 16) & 1u);   // round-to-nearest-even
  return (unsigned short)(u >> 16);
}

union FragU { v4u u[2]; v16bf b; };

// A fragment (16x32 bf16): lane m=lane&15; lo-half lanes K=kk+{0..15},
// hi-half lanes K=kk+{16..31}.  Two aligned 16B LDS loads.
__device__ __forceinline__ v16bf load_frag_A(const unsigned short* base, int ldk,
                                             int kk, int lane) {
  int m  = lane & 15;
  int hi = lane >> 4;
  const unsigned short* p = base + m * ldk + kk + hi * 16;
  FragU f;
  f.u[0] = *(const v4u*)(p);
  f.u[1] = *(const v4u*)(p + 8);
  return f.b;
}

// B fragment (32x16 bf16): lane n=n0+(lane&15); K = kk + (lane>=16?16:0) + j.
// 16 consecutive halves -> two aligned 16B global loads.
__device__ __forceinline__ v16bf load_frag_B(const unsigned short* base, int ldk,
                                             int n0, int kk, int lane) {
  int n  = n0 + (lane & 15);
  int kh = lane >> 4;
  const unsigned short* p = base + n * ldk + kk + kh * 16;
  FragU f;
  f.u[0] = *(const v4u*)(p);
  f.u[1] = *(const v4u*)(p + 8);
  return f.b;
}

__device__ __forceinline__ v8f wmma_bf16(v16bf a, v16bf b, v8f c) {
  return __builtin_amdgcn_wmma_f32_16x16x32_bf16(false, a, false, b,
                                                 (short)0, c, false, false);
}

__device__ __forceinline__ v8f zero8() {
  v8f z;
  #pragma unroll
  for (int i = 0; i < 8; ++i) z[i] = 0.f;
  return z;
}

__device__ __forceinline__ float sigmoidf(float x) {
  return 1.f / (1.f + expf(-x));
}

// ---------------------------------------------------------------------------
// Prep: convert fp32 weights -> padded bf16 panels in d_ws (L2-resident).
// ---------------------------------------------------------------------------
__global__ __launch_bounds__(256) void dyn_prep(const float* __restrict__ W_ih,
                                                const float* __restrict__ W_hh,
                                                const float* __restrict__ W1,
                                                const float* __restrict__ W2,
                                                unsigned short* __restrict__ ws) {
  int idx = blockIdx.x * 256 + threadIdx.x;
  if (idx >= WTOT) return;
  float v = 0.f;
  if (idx < OFF_NIH) {                       // fused r/z panels [160][192]
    int g = idx / 30720;
    int j = idx - g * 30720;
    int i = j / 192, k = j - i * 192;
    if (i < DG) {
      int row = g * DG + i;
      if (k < 32) { if (k < DIN) v = W_ih[row * DIN + k]; }
      else        { int kh = k - 32; if (kh < DG) v = W_hh[row * DG + kh]; }
    }
  } else if (idx < OFF_NHH) {                // n gate, W_ih part [160][32]
    int j = idx - OFF_NIH;
    int i = j / 32, k = j - i * 32;
    if (i < DG && k < DIN) v = W_ih[(2 * DG + i) * DIN + k];
  } else if (idx < OFF_W1) {                 // n gate, W_hh part [160][160]
    int j = idx - OFF_NHH;
    int i = j / 160, k = j - i * 160;
    if (i < DG && k < DG) v = W_hh[(2 * DG + i) * DG + k];
  } else if (idx < OFF_W2) {                 // W1 [64][224]
    int j = idx - OFF_W1;
    int i = j / 224, k = j - i * 224;
    if (i < 50 && k < 200) v = W1[i * 200 + k];
  } else {                                   // W2 [64][64]
    int j = idx - OFF_W2;
    int i = j / 64, k = j - i * 64;
    if (i < 50 && k < 50) v = W2[i * 50 + k];
  }
  ws[idx] = f2bf(v);
}

// ---------------------------------------------------------------------------
// Main: 64 agents per block, 8 wave32, all GEMMs on v_wmma_f32_16x16x32_bf16.
// ---------------------------------------------------------------------------
__global__ __launch_bounds__(256) void dyn_main(
    const float* __restrict__ z,     const float* __restrict__ x_flat,
    const float* __restrict__ h,     const float* __restrict__ statc,
    const float* __restrict__ dx,    const float* __restrict__ x_prev,
    const float* __restrict__ b_ih,  const float* __restrict__ b_hh,
    const float* __restrict__ b1,    const float* __restrict__ b2,
    const float* __restrict__ W3,    const float* __restrict__ b3,
    const unsigned short* __restrict__ ws,
    float* __restrict__ out, int NA) {
  extern __shared__ __align__(16) char smem[];
  unsigned short* sA1 = (unsigned short*)(smem + SA1_B);
  unsigned short* sA2 = (unsigned short*)(smem + SA2_B);
  unsigned short* sA3 = (unsigned short*)(smem + SA3_B);
  float*          sH  = (float*)(smem + SH_B);
  float*          sO6 = (float*)(smem + SO6_B);

  const int tid  = threadIdx.x;
  const int lane = tid & 31;
  const int wave = tid >> 5;
  const int aBase = blockIdx.x * MTILE;

  const int X_BASE = 0, MU_BASE = 2 * NA, SIG_BASE = 4 * NA, H_BASE = 8 * NA;

  // ---- Stage 0: stage A1 = [x_flat(0..31 pad) | h(32..191 pad)] as bf16 ----
  for (int idx = tid; idx < MTILE * 192; idx += 256) {
    int m = idx / 192, k = idx - m * 192;
    int a = aBase + m;
    float v = 0.f;
    if (a < NA) {
      if (k < 32) { if (k < DIN) v = x_flat[a * DIN + k]; }
      else        { int kh = k - 32; if (kh < DG) v = h[a * DG + kh]; }
    }
    sA1[m * 192 + k] = f2bf(v);
  }
  // A2 static tail (cols 150..223); h_new part filled after the GRU GEMM.
  for (int idx = tid; idx < MTILE * 74; idx += 256) {
    int m = idx / 74, k = 150 + (idx - m * 74);
    int a = aBase + m;
    float v = (a < NA && k < DG + DS) ? statc[a * DS + (k - DG)] : 0.f;
    sA2[m * 224 + k] = f2bf(v);
  }
  __syncthreads();

  // ---- GEMM1 (GRU gates) + in-register gate math --------------------------
  // 10 i-tiles (gate dim 150 pad 160) x 4 m-tiles = 40 jobs / 8 waves.
  for (int p = wave; p < 40; p += 8) {
    int t = p >> 2, mi = p & 3;
    const unsigned short* A = sA1 + mi * 16 * 192;
    v8f ar = zero8(), az = zero8(), axn = zero8(), ahn = zero8();
    #pragma unroll
    for (int ks = 0; ks < 6; ++ks) {
      int kk = ks * 32;
      v16bf a = load_frag_A(A, 192, kk, lane);
      v16bf br = load_frag_B(ws + OFF_R, 192, t * 16, kk, lane);
      ar = wmma_bf16(a, br, ar);                       // xr + hr
      v16bf bz = load_frag_B(ws + OFF_Z, 192, t * 16, kk, lane);
      az = wmma_bf16(a, bz, az);                       // xz + hz
      if (ks == 0) {
        v16bf bn = load_frag_B(ws + OFF_NIH, 32, t * 16, 0, lane);
        axn = wmma_bf16(a, bn, axn);                   // xn
      } else {
        v16bf bn = load_frag_B(ws + OFF_NHH, 160, t * 16, kk - 32, lane);
        ahn = wmma_bf16(a, bn, ahn);                   // hn
      }
    }
    // Gate math directly on the WMMA C layout: M = mi*16 + e + 8*(lane>=16),
    // N(=gate index i) = t*16 + (lane&15).
    int i = t * 16 + (lane & 15);
    if (i < DG) {
      int hiL = lane >> 4;
      float bias_r = b_ih[i] + b_hh[i];
      float bias_z = b_ih[DG + i] + b_hh[DG + i];
      float bn_x = b_ih[2 * DG + i];
      float bn_h = b_hh[2 * DG + i];
      #pragma unroll
      for (int e = 0; e < 8; ++e) {
        int m = mi * 16 + hiL * 8 + e;
        int a = aBase + m;
        if (a < NA) {
          float r  = sigmoidf(ar[e] + bias_r);
          float zg = sigmoidf(az[e] + bias_z);
          float n  = tanhf(axn[e] + bn_x + r * (ahn[e] + bn_h));
          float h0 = h[a * DG + i];                    // exact fp32 blend
          float hn = (1.f - zg) * n + zg * h0;
          out[H_BASE + a * DG + i] = hn;               // h_new output
          sA2[m * 224 + i] = f2bf(hn);                 // feed MLP layer 1
        }
      }
    }
  }
  __syncthreads();

  // ---- GEMM2: [64x224] x W1 -> softplus -> sA3 bf16 -----------------------
  for (int p = wave; p < 16; p += 8) {
    int mi = p >> 2, tn = p & 3;
    const unsigned short* A = sA2 + mi * 16 * 224;
    v8f acc = zero8();
    #pragma unroll
    for (int ks = 0; ks < 7; ++ks) {
      v16bf a = load_frag_A(A, 224, ks * 32, lane);
      v16bf b = load_frag_B(ws + OFF_W1, 224, tn * 16, ks * 32, lane);
      acc = wmma_bf16(a, b, acc);
    }
    int nn = tn * 16 + (lane & 15);
    int hiL = lane >> 4;
    float bb = (nn < 50) ? b1[nn] : 0.f;
    #pragma unroll
    for (int e = 0; e < 8; ++e) {
      int m = mi * 16 + hiL * 8 + e;
      float v = acc[e] + bb;
      float sp = (v > 20.f) ? v : log1pf(expf(v));     // stable softplus
      sA3[m * 64 + nn] = (nn < 50) ? f2bf(sp) : (unsigned short)0;
    }
  }
  __syncthreads();

  // ---- GEMM3: [64x64] x W2 -> tanh -> sH f32 ------------------------------
  for (int p = wave; p < 16; p += 8) {
    int mi = p >> 2, tn = p & 3;
    const unsigned short* A = sA3 + mi * 16 * 64;
    v8f acc = zero8();
    #pragma unroll
    for (int ks = 0; ks < 2; ++ks) {
      v16bf a = load_frag_A(A, 64, ks * 32, lane);
      v16bf b = load_frag_B(ws + OFF_W2, 64, tn * 16, ks * 32, lane);
      acc = wmma_bf16(a, b, acc);
    }
    int nn = tn * 16 + (lane & 15);
    int hiL = lane >> 4;
    float bb = (nn < 50) ? b2[nn] : 0.f;
    #pragma unroll
    for (int e = 0; e < 8; ++e) {
      int m = mi * 16 + hiL * 8 + e;
      sH[m * 64 + nn] = tanhf(acc[e] + bb);
    }
  }
  __syncthreads();

  // ---- Layer 3: 6x50, scalar (negligible FLOPs) ---------------------------
  for (int idx = tid; idx < MTILE * 6; idx += 256) {
    int m = idx / 6, o = idx - m * 6;
    float acc = b3[o];
    #pragma unroll 5
    for (int j = 0; j < 50; ++j) acc += sH[m * 64 + j] * W3[o * 50 + j];
    sO6[m * 8 + o] = acc;
  }
  __syncthreads();

  // ---- Head math: mu / sigma / x, one thread per agent --------------------
  if (tid < MTILE) {
    int m = tid;
    int a = aBase + m;
    if (a < NA) {
      float o0 = sO6[m * 8 + 0], o1 = sO6[m * 8 + 1];
      float o2 = sO6[m * 8 + 2], o3 = sO6[m * 8 + 3];
      float o4 = sO6[m * 8 + 4], o5 = sO6[m * 8 + 5];
      float mu0 = x_prev[a * 2 + 0] + 0.5f * dx[a * 2 + 0] + o0;
      float mu1 = x_prev[a * 2 + 1] + 0.5f * dx[a * 2 + 1] + o1;
      float bq = o3 + o4;
      float apd = o2 + o5;
      float amd = o2 - o5;
      float delta = sqrtf(amd * amd + bq * bq);
      float e = expf(delta), inv = 1.f / e;
      float sh = 0.5f * (e - inv), ch = 0.5f * (e + inv);
      float tmp1 = sh / delta;
      float tmp2 = amd / tmp1;
      float sc = expf(apd);
      float s00 = (ch + tmp2) * sc;
      float s01 = (bq * tmp1) * sc;
      float s11 = (ch - tmp2) * sc;
      float z0 = z[a * 2 + 0], z1 = z[a * 2 + 1];
      out[X_BASE + a * 2 + 0] = s00 * z0 + s01 * z1 + mu0;
      out[X_BASE + a * 2 + 1] = s01 * z0 + s11 * z1 + mu1;
      out[MU_BASE + a * 2 + 0] = mu0;
      out[MU_BASE + a * 2 + 1] = mu1;
      out[SIG_BASE + a * 4 + 0] = s00;
      out[SIG_BASE + a * 4 + 1] = s01;
      out[SIG_BASE + a * 4 + 2] = s01;
      out[SIG_BASE + a * 4 + 3] = s11;
    }
  }
}

// ---------------------------------------------------------------------------
extern "C" void kernel_launch(void* const* d_in, const int* in_sizes, int n_in,
                              void* d_out, int out_size, void* d_ws, size_t ws_size,
                              hipStream_t stream) {
  const float* z      = (const float*)d_in[0];
  const float* x_flat = (const float*)d_in[1];
  const float* h      = (const float*)d_in[2];
  const float* statc  = (const float*)d_in[3];
  const float* dx     = (const float*)d_in[4];
  const float* x_prev = (const float*)d_in[5];
  const float* W_ih   = (const float*)d_in[6];
  const float* W_hh   = (const float*)d_in[7];
  const float* b_ih   = (const float*)d_in[8];
  const float* b_hh   = (const float*)d_in[9];
  const float* W1     = (const float*)d_in[10];
  const float* b1     = (const float*)d_in[11];
  const float* W2     = (const float*)d_in[12];
  const float* b2     = (const float*)d_in[13];
  const float* W3     = (const float*)d_in[14];
  const float* b3     = (const float*)d_in[15];
  float* out = (float*)d_out;
  unsigned short* ws = (unsigned short*)d_ws;

  const int NA = in_sizes[0] / 2;

  dyn_prep<<<(WTOT + 255) / 256, 256, 0, stream>>>(W_ih, W_hh, W1, W2, ws);

  const int grid = (NA + MTILE - 1) / MTILE;
  dyn_main<<<grid, 256, SMEM_BYTES, stream>>>(z, x_flat, h, statc, dx, x_prev,
                                              b_ih, b_hh, b1, b2, W3, b3,
                                              ws, out, NA);
}
